// SE3Net_76630806495656
// MI455X (gfx1250) — matrix operand order repaired and checked by reading, not compiled, not run on hardware
//
#include <hip/hip_runtime.h>

// ---------------- problem constants ----------------
#define Bn    4
#define Nn    286
#define CIN   23
#define COUT  2
#define Hh    100
#define NLAY  49            // hidden->hidden layers (L-1)
#define PTOT  (Bn*Nn*Nn)    // 327184 atom pairs
#define MTILE 128           // pairs per workgroup
#define NWG   ((PTOT + MTILE - 1) / MTILE)

#define KT 4                // K tiles: 128 / 32
#define NT 7                // N tiles (hidden): 112 / 16
#define MT 8                // M tiles: 128 / 16
#define NT_OUT 3            // N tiles (output): 48 / 16

#define WSTRIDE (KT*NT*32*16)     // 14336 bf16 elements per packed layer
#define WBYTES  (WSTRIDE*2)       // 28672 bytes per packed layer
#define NPACK   (NLAY + 1)        // 49 hidden layers + output layer

typedef __attribute__((ext_vector_type(16))) __bf16 v16bf;
typedef __attribute__((ext_vector_type(8)))  float  v8f;

__device__ __forceinline__ __bf16 f2bf(float f) {
  unsigned u = __builtin_bit_cast(unsigned, f);
  u += 0x7FFFu + ((u >> 16) & 1u);           // round-to-nearest-even
  unsigned short s = (unsigned short)(u >> 16);
  return __builtin_bit_cast(__bf16, s);
}

// ---- async global -> LDS staging of one packed weight layer (28672 B) ----
// 256 threads x 7 x b128. Tracked by ASYNCcnt; completes independently of
// the normal load/ds counters, so it overlaps the activation write-back.
__device__ __forceinline__ void stage_layer(const __bf16* __restrict__ wpack,
                                            int l, void* bpack, int tid) {
#if defined(__gfx1250__)
  unsigned long long ga =
      (unsigned long long)((const char*)wpack + (size_t)l * WBYTES + tid * 16);
  // low 32 bits of a generic pointer to LDS == byte offset in wave's LDS
  unsigned lds = (unsigned)(unsigned long long)bpack + (unsigned)(tid * 16);
  #pragma unroll
  for (int i = 0; i < 7; ++i) {
    asm volatile("global_load_async_to_lds_b128 %0, %1, off"
                 :: "v"(lds + (unsigned)(i * 4096)),
                    "v"(ga + (unsigned long long)(i * 4096))
                 : "memory");
  }
#else
  // host pass / non-gfx1250: synchronous copy
  struct B16 { unsigned a, b, c, d; };
  const B16* s = (const B16*)((const char*)wpack + (size_t)l * WBYTES);
  B16* d = (B16*)bpack;
  for (int i = 0; i < 7; ++i) d[tid + i * 256] = s[tid + i * 256];
#endif
}

__device__ __forceinline__ void wait_async0() {
#if defined(__gfx1250__)
  asm volatile("s_wait_asynccnt 0" ::: "memory");
#endif
}

// ---------------- one-time weight prepack ----------------
// w_hid[49][100][100] (*0.1) and w_out[100][46] (*0.1) -> bf16 in the exact
// WMMA B-fragment-packed LDS image: [(kt*NT+nt)*32 + lane]*16 + e, where
// k = kt*32 + (e>>3)*16 + (lane>>4)*8 + (e&7), n = nt*16 + (lane&15). Zero padding built in.
__global__ __launch_bounds__(256)
void prepack_weights(const float* __restrict__ w_hid,
                     const float* __restrict__ w_out,
                     __bf16* __restrict__ wpack) {
  int idx = blockIdx.x * 256 + threadIdx.x;
  if (idx >= NPACK * WSTRIDE) return;
  int l = idx / WSTRIDE;
  int r = idx - l * WSTRIDE;
  int e    = r & 15;
  int ln   = (r >> 4) & 31;
  int tile = r >> 9;                   // kt*NT + nt
  int kt = tile / NT, nt = tile - kt * NT;
  int k = kt * 32 + (e >> 3) * 16 + (ln >> 4) * 8 + (e & 7);
  int n = nt * 16 + (ln & 15);
  float v = 0.f;
  if (l < NLAY) {
    if (k < Hh && n < Hh) v = w_hid[l * Hh * Hh + k * Hh + n] * 0.1f;
  } else {
    if (k < Hh && n < COUT * CIN) v = w_out[k * (COUT * CIN) + n] * 0.1f;
  }
  wpack[idx] = f2bf(v);
}

// ---------------- main fused kernel ----------------
__global__ __launch_bounds__(256)
void se3_mlp_conv(const float* __restrict__ feat,
                  const float* __restrict__ geom,
                  const float* __restrict__ w_in,
                  const __bf16* __restrict__ wpack,
                  float* __restrict__ out_pre) {
  __shared__ v16bf Apack[MT * KT * 32];   // 32 KB: activations, frag-packed
  __shared__ v16bf Bpack[KT * NT * 32];   // 28 KB: weights, frag-packed
  __shared__ float basisS[MTILE][4];      //  2 KB

  const int tid  = threadIdx.x;
  const int lane = tid & 31;
  const int wave = tid >> 5;
  const int p0   = blockIdx.x * MTILE;

  __bf16* Ab = (__bf16*)Apack;

  // kick off async DMA of layer-0 weights immediately
  stage_layer(wpack, 0, Bpack, tid);

  // ---- cosine-bump basis per pair (mu = 0,1.5,3.0; step 1.5) ----
  if (tid < MTILE) {
    int p = p0 + tid;
    float b3[3] = {0.f, 0.f, 0.f};
    if (p < PTOT) {
      int b   = p / (Nn * Nn);
      int rem = p - b * (Nn * Nn);
      int i   = rem / Nn;
      int j   = rem - i * Nn;
      const float* gi = geom + (b * Nn + i) * 3;
      const float* gj = geom + (b * Nn + j) * 3;
      float dx = gi[0] - gj[0], dy = gi[1] - gj[1], dz = gi[2] - gj[2];
      float r = sqrtf(dx * dx + dy * dy + dz * dz + 1e-12f);
      #pragma unroll
      for (int k = 0; k < 3; ++k) {
        float x = (r - 1.5f * (float)k) * (1.0f / 1.5f);
        float v = 0.f;
        if (fabsf(x) < 1.f) {
          float c = __cosf(1.57079632679489662f * x);
          v = c * c;
        }
        b3[k] = v;
      }
    }
    basisS[tid][0] = b3[0]; basisS[tid][1] = b3[1]; basisS[tid][2] = b3[2];
  }
  __syncthreads();

  // ---- input layer: h0 = relu(basis @ w_in/sqrt(3)) -> Apack (frag-packed) ----
  {
    const float is3 = 0.57735026918962576f;
    for (int idx = tid; idx < MTILE * 128; idx += 256) {   // 64 iters, uniform
      int m = idx >> 7, k = idx & 127;
      float v = 0.f;
      if (k < Hh) {
        float s = basisS[m][0] * w_in[0 * Hh + k]
                + basisS[m][1] * w_in[1 * Hh + k]
                + basisS[m][2] * w_in[2 * Hh + k];
        v = fmaxf(s * is3, 0.f);
      }
      int mt = m >> 4, mr = m & 15;
      int kt = k >> 5, kr = k & 31;
      int h  = (kr & 15) >> 3;
      int e  = (kr & 7) | ((kr >> 4) << 3);
      Ab[(((mt * KT) + kt) * 32 + h * 16 + mr) * 16 + e] = f2bf(v);
    }
  }

  // ---- 49 hidden layers: h = relu(h @ W/10) via WMMA bf16 ----
  for (int l = 0; l < NLAY; ++l) {
    wait_async0();        // this wave's Bpack DMA complete
    __syncthreads();      // everyone's DMA + previous write-back visible

    if (l + 2 <= NLAY)    // warm L2 for layer l+2 (global_prefetch_b8)
      __builtin_prefetch((const char*)wpack + (size_t)(l + 2) * WBYTES + tid * 112, 0, 1);

    // compute: wave owns M-tile `wave`, all 7 N-tiles; accs stay in VGPRs
    const int mt = wave;
    v16bf afr[KT];
    #pragma unroll
    for (int kt = 0; kt < KT; ++kt) afr[kt] = Apack[(mt * KT + kt) * 32 + lane];

    v8f acc[NT];
    #pragma unroll
    for (int nt = 0; nt < NT; ++nt) {
      v8f a = {};
      #pragma unroll
      for (int kt = 0; kt < KT; ++kt) {
        v16bf bfr = Bpack[(kt * NT + nt) * 32 + lane];
        a = __builtin_amdgcn_wmma_f32_16x16x32_bf16(
                false, afr[kt], false, bfr, (short)0, a, false, false);
      }
      acc[nt] = a;
    }
    __syncthreads();   // all reads of Apack/Bpack done before overwrite

    // overlap: DMA next layer's weights while we write activations back
    stage_layer(wpack, l + 1, Bpack, tid);

    // relu + bf16, scatter back into Apack (next layer's A, frag-packed)
    const int hl = lane >> 4, nl = lane & 15;
    #pragma unroll
    for (int nt = 0; nt < NT; ++nt) {
      int kcol = nt * 16 + nl;           // becomes next layer's K coordinate
      int kt2 = kcol >> 5, kr2 = kcol & 31;
      int h2 = (kr2 & 15) >> 3;
      int e2 = (kr2 & 7) | ((kr2 >> 4) << 3);
      #pragma unroll
      for (int v = 0; v < 8; ++v) {
        int mloc = hl * 8 + v;           // row within M-tile
        float val = fmaxf(acc[nt][v], 0.f);
        Ab[(((mt * KT) + kt2) * 32 + h2 * 16 + mloc) * 16 + e2] = f2bf(val);
      }
    }
    // no extra barrier: next iteration's top barrier orders everything
  }

  // ---- output layer: R = h @ w_out/10, then per-pair einsum contribution ----
  wait_async0();          // packed w_out (layer NLAY) DMA complete
  __syncthreads();

  {
    const int mt = wave;
    const int hl = lane >> 4, nl = lane & 15;
    const float OUT_SCALE = 0.14104739588693907f;   // Y00 / sqrt(4)
    v16bf afr[KT];
    #pragma unroll
    for (int kt = 0; kt < KT; ++kt) afr[kt] = Apack[(mt * KT + kt) * 32 + lane];

    #pragma unroll
    for (int nt = 0; nt < NT_OUT; ++nt) {
      v8f acc = {};
      #pragma unroll
      for (int kt = 0; kt < KT; ++kt) {
        v16bf bfr = Bpack[(kt * NT + nt) * 32 + lane];
        acc = __builtin_amdgcn_wmma_f32_16x16x32_bf16(
                  false, afr[kt], false, bfr, (short)0, acc, false, false);
      }
      int col = nt * 16 + nl;           // flat (o, c) index, valid < 46
      if (col < COUT * CIN) {
        int o = col / CIN, c = col - o * CIN;
        #pragma unroll
        for (int v = 0; v < 8; ++v) {
          int p = p0 + mt * 16 + hl * 8 + v;
          if (p < PTOT) {
            int b   = p / (Nn * Nn);
            int rem = p - b * (Nn * Nn);
            int i   = rem / Nn;
            int j   = rem - i * Nn;
            float contrib = acc[v] * OUT_SCALE * feat[(b * Nn + j) * CIN + c];
            atomicAdd(&out_pre[(b * Nn + i) * COUT + o], contrib);
          }
        }
      }
    }
  }
}

// ---------------- helpers ----------------
__global__ void zero_kernel(float* p, int n) {
  int i = blockIdx.x * blockDim.x + threadIdx.x;
  if (i < n) p[i] = 0.f;
}

// relu -> mean over atoms -> linear head ; one block per batch element
__global__ __launch_bounds__(256)
void finalize_kernel(const float* __restrict__ pre,
                     const float* __restrict__ lin_w,
                     const float* __restrict__ lin_b,
                     float* __restrict__ out) {
  __shared__ float r0[256], r1[256];
  int b = blockIdx.x, tid = threadIdx.x;
  float s0 = 0.f, s1 = 0.f;
  for (int i = tid; i < Nn; i += 256) {
    s0 += fmaxf(pre[(b * Nn + i) * COUT + 0], 0.f);
    s1 += fmaxf(pre[(b * Nn + i) * COUT + 1], 0.f);
  }
  r0[tid] = s0; r1[tid] = s1;
  __syncthreads();
  for (int off = 128; off > 0; off >>= 1) {
    if (tid < off) { r0[tid] += r0[tid + off]; r1[tid] += r1[tid + off]; }
    __syncthreads();
  }
  if (tid == 0) {
    float inv = 1.0f / (float)Nn;
    out[b] = (r0[0] * inv) * lin_w[0] + (r1[0] * inv) * lin_w[1] + lin_b[0];
  }
}

// ---------------- entry point ----------------
extern "C" void kernel_launch(void* const* d_in, const int* in_sizes, int n_in,
                              void* d_out, int out_size, void* d_ws, size_t ws_size,
                              hipStream_t stream) {
  (void)in_sizes; (void)n_in; (void)out_size; (void)ws_size;
  const float* feat  = (const float*)d_in[0];   // [B,N,CIN]
  const float* geom  = (const float*)d_in[1];   // [B,N,3]
  // d_in[2] = num_atoms (constant N, unused)
  const float* w_in  = (const float*)d_in[3];   // [3,H]
  const float* w_hid = (const float*)d_in[4];   // [49,H,H]
  const float* w_out = (const float*)d_in[5];   // [H,46]
  const float* lin_w = (const float*)d_in[6];   // [2,1]
  const float* lin_b = (const float*)d_in[7];   // [1]
  float* out     = (float*)d_out;               // [B]

  // workspace layout: [0, 9152): out_pre accumulators ; [16384, +1.4MB): wpack
  float*  out_pre = (float*)d_ws;
  __bf16* wpack   = (__bf16*)((char*)d_ws + 16384);

  const int npre   = Bn * Nn * COUT;
  const int ptotal = NPACK * WSTRIDE;

  prepack_weights<<<(ptotal + 255) / 256, 256, 0, stream>>>(w_hid, w_out, wpack);
  zero_kernel<<<(npre + 255) / 256, 256, 0, stream>>>(out_pre, npre);
  se3_mlp_conv<<<NWG, 256, 0, stream>>>(feat, geom, w_in, wpack, out_pre);
  finalize_kernel<<<Bn, 256, 0, stream>>>(out_pre, lin_w, lin_b, out);
}
